// HGT_7868380086415
// MI455X (gfx1250) — matrix operand order repaired and checked by reading, not compiled, not run on hardware
//
#include <hip/hip_runtime.h>
#include <math.h>

#define NNODES 100000
#define NEDGES 1000000
#define NEGV  -1e30f
#define GLOBAL __attribute__((address_space(1)))

typedef __attribute__((ext_vector_type(16))) _Float16 v16h;
typedef __attribute__((ext_vector_type(8)))  _Float16 v8h;
typedef __attribute__((ext_vector_type(8)))  float    v8f;
typedef __attribute__((ext_vector_type(4)))  float    v4f;

// ---- global-address-space access helpers (force global_load/_store, not flat) ----
static __device__ __forceinline__ v8h ldg_v8h(const _Float16* p) {
  return *(const GLOBAL v8h*)(unsigned long long)p;
}
static __device__ __forceinline__ v4f ldg_v4f(const float* p) {
  return *(const GLOBAL v4f*)(unsigned long long)p;
}
static __device__ __forceinline__ float ldg_f(const float* p) {
  return *(const GLOBAL float*)(unsigned long long)p;
}
static __device__ __forceinline__ int ldg_i(const int* p) {
  return *(const GLOBAL int*)(unsigned long long)p;
}
static __device__ __forceinline__ void stg_f(float* p, float v) {
  *(GLOBAL float*)(unsigned long long)p = v;
}
static __device__ __forceinline__ void stg_h(_Float16* p, _Float16 v) {
  *(GLOBAL _Float16*)(unsigned long long)p = v;
}

static __device__ __forceinline__ v8f wmma16(v16h a, v16h b, v8f c) {
  // D = A(16x32 f16) x B(32x16 f16) + C(16x16 f32)
  return __builtin_amdgcn_wmma_f32_16x16x32_f16(false, a, false, b, (short)0, c, false, false);
}
static __device__ __forceinline__ v16h cat8(v8h lo, v8h hi) {
  return __builtin_shufflevector(lo, hi, 0,1,2,3,4,5,6,7,8,9,10,11,12,13,14,15);
}
static __device__ __forceinline__ v8f vzero() {
  v8f z = {0.f,0.f,0.f,0.f,0.f,0.f,0.f,0.f}; return z;
}

// A operand (16x32 f16): lane row m=l%16, hi=l/16.
// element j<8 : A[m][kb+hi*8+j] ; j>=8 : A[m][kb+16+hi*8+(j-8)]  (ISA 16-bit A layout)
static __device__ __forceinline__ v16h loadA_g(const _Float16* row, int kb, int hi) {
  return cat8(ldg_v8h(row + kb + hi*8), ldg_v8h(row + kb + 16 + hi*8));
}
static __device__ __forceinline__ v16h loadA_l(const _Float16* row, int kb, int hi) {
  const v8h* p0 = (const v8h*)(row + kb + hi*8);        // LDS: keep generic -> ds ops
  const v8h* p1 = (const v8h*)(row + kb + 16 + hi*8);
  return cat8(*p0, *p1);
}
static __device__ __forceinline__ v16h loadA_f(const float* row, int kb, int hi) {
  v4f a0 = ldg_v4f(row + kb + hi*8),      a1 = ldg_v4f(row + kb + hi*8 + 4);
  v4f b0 = ldg_v4f(row + kb + 16 + hi*8), b1 = ldg_v4f(row + kb + 16 + hi*8 + 4);
  v16h r;
#pragma unroll
  for (int j = 0; j < 4; ++j) {
    r[j]    = (_Float16)a0[j]; r[4+j]  = (_Float16)a1[j];
    r[8+j]  = (_Float16)b0[j]; r[12+j] = (_Float16)b1[j];
  }
  return r;
}
// B operand (32x16 f16): lane col n=l%16, hi=l/16. element j<8 : WT[n][kb+hi*8+j] ; j>=8 : WT[n][kb+16+hi*8+(j-8)]
// (same k order as A; WT is [N][K] transposed)
static __device__ __forceinline__ v16h loadB_h(const _Float16* wrow, int kb, int hi) {
  return cat8(ldg_v8h(wrow + kb + hi*8), ldg_v8h(wrow + kb + 16 + hi*8));
}
static __device__ __forceinline__ void atomicMaxF(float* addr, float val) {
  if (val >= 0.f) atomicMax((int*)addr, __float_as_int(val));
  else            atomicMin((unsigned int*)addr, __float_as_uint(val));
}

// ---------------------------------------------------------------- weight prep
struct PrepArgs {
  const float *W1, *W2, *W3, *Wc, *Wp;
  const float *Wq[2], *Wk[2], *Wv[2], *Wa[2];
  const float *ar[2], *mr[2], *pr[2], *bk[2], *bv[2];
  _Float16 *W1T, *W2T, *W3T, *WCT, *WPT;
  _Float16 *WQT[2], *WAT[2], *WKT[2][2], *WVT[2][2];
  float *BKT[2][2], *BVT[2][2];
};

static __device__ void transposeW(const float* src, _Float16* dst, int K, int tid, int nt) {
  for (int i = tid; i < 64 * K; i += nt) {
    int n = i / K, k = i - n * K;
    stg_h(dst + i, (_Float16)ldg_f(src + (size_t)k * 64 + n));
  }
}

__global__ __launch_bounds__(256) void k_prep(PrepArgs a) {
  const int tid = threadIdx.x, nt = blockDim.x;
  switch (blockIdx.x) {
    case 0: transposeW(a.W1, a.W1T, 768, tid, nt); break;
    case 1: transposeW(a.W2, a.W2T, 768, tid, nt); break;
    case 2: transposeW(a.W3, a.W3T, 128, tid, nt); break;
    case 3: transposeW(a.Wc, a.WCT, 192, tid, nt); break;
    case 6: transposeW(a.Wp, a.WPT,  64, tid, nt); break;
    case 4: case 5: {
      const int L = blockIdx.x - 4;
      transposeW(a.Wq[L], a.WQT[L], 64, tid, nt);
      transposeW(a.Wa[L], a.WAT[L], 64, tid, nt);
      for (int t = 0; t < 2; ++t) {
        // combined: WKT[o=h*16+e][in] = sum_d Wk[in][h*16+d]*a_rel[t,h,d,e]*p_rel[t,h]*scale
        for (int i = tid; i < 64 * 64; i += nt) {
          int o = i >> 6, in = i & 63, h = o >> 4, e = o & 15;
          float s = ldg_f(a.pr[L] + t*4 + h) * 0.25f;   // scale = 1/sqrt(16)
          float vk = 0.f, vv = 0.f;
          for (int d = 0; d < 16; ++d) {
            int ai = ((t*4 + h)*16 + d)*16 + e;
            vk += ldg_f(a.Wk[L] + (size_t)in*64 + h*16 + d) * ldg_f(a.ar[L] + ai);
            vv += ldg_f(a.Wv[L] + (size_t)in*64 + h*16 + d) * ldg_f(a.mr[L] + ai);
          }
          stg_h(a.WKT[L][t] + (size_t)o*64 + in, (_Float16)(vk * s));
          stg_h(a.WVT[L][t] + (size_t)o*64 + in, (_Float16)vv);
        }
        for (int o = tid; o < 64; o += nt) {
          int h = o >> 4, e = o & 15;
          float s = ldg_f(a.pr[L] + t*4 + h) * 0.25f;
          float bkv = 0.f, bvv = 0.f;
          for (int d = 0; d < 16; ++d) {
            int ai = ((t*4 + h)*16 + d)*16 + e;
            bkv += ldg_f(a.bk[L] + h*16 + d) * ldg_f(a.ar[L] + ai);
            bvv += ldg_f(a.bv[L] + h*16 + d) * ldg_f(a.mr[L] + ai);
          }
          stg_f(a.BKT[L][t] + o, bkv * s);
          stg_f(a.BVT[L][t] + o, bvv);
        }
      }
      break;
    }
    default: break;
  }
}

// ---------------------------------------------------------------- fused input MLP
struct MlpArgs {
  const float *x1, *x2, *x3;
  const _Float16 *W1T, *W2T, *W3T, *WCT;
  const float *b1, *b2, *b3, *bc;
  float* xf32; _Float16* xf16;
};

__global__ __launch_bounds__(256) void k_mlp(MlpArgs A) {
  __shared__ _Float16 sh[8][16][72];
  const int lane = threadIdx.x & 31, wv = threadIdx.x >> 5;
  const int m = lane & 15, hi = lane >> 4;
  const int tile = blockIdx.x * 8 + wv;
  const bool act = (tile * 16) < NNODES;
  const int row0 = act ? tile * 16 : 0;

  const float*    xs[3] = {A.x1, A.x2, A.x3};
  const _Float16* WT[3] = {A.W1T, A.W2T, A.W3T};
  const float*    bs[3] = {A.b1, A.b2, A.b3};
  const int       Ks[3] = {768, 768, 128};

  v8f accC[4] = {vzero(), vzero(), vzero(), vzero()};

  for (int p = 0; p < 3; ++p) {
    const int K = Ks[p];
    v8f acc[4] = {vzero(), vzero(), vzero(), vzero()};
    const float* arow = xs[p] + (size_t)(row0 + m) * K;
    for (int kb = 0; kb < K; kb += 32) {
      if (kb + 64 <= K) __builtin_prefetch(arow + kb + 64, 0, 1);  // global_prefetch_b8
      v16h a = loadA_f(arow, kb, hi);
#pragma unroll
      for (int ntile = 0; ntile < 4; ++ntile) {
        v16h b = loadB_h(WT[p] + (size_t)(ntile*16 + m) * K, kb, hi);
        acc[ntile] = wmma16(a, b, acc[ntile]);
      }
    }
    __syncthreads();                 // previous Wc-GEMM reads of sh are done
#pragma unroll
    for (int ntile = 0; ntile < 4; ++ntile)
#pragma unroll
      for (int r = 0; r < 8; ++r) {  // C layout: row = r + hi*8, col = m + ntile*16
        int rr = r + hi*8, cc = m + ntile*16;
        float v = acc[ntile][r] + ldg_f(bs[p] + cc);
        sh[wv][rr][cc] = (_Float16)(v > 0.f ? v : 0.f);
      }
    __syncthreads();
#pragma unroll
    for (int kb = 0; kb < 64; kb += 32) {
      v16h a = loadA_l(&sh[wv][m][0], kb, hi);
#pragma unroll
      for (int ntile = 0; ntile < 4; ++ntile) {
        v16h b = loadB_h(A.WCT + (size_t)(ntile*16 + m) * 192, p*64 + kb, hi);
        accC[ntile] = wmma16(a, b, accC[ntile]);
      }
    }
  }
  if (act) {
#pragma unroll
    for (int ntile = 0; ntile < 4; ++ntile)
#pragma unroll
      for (int r = 0; r < 8; ++r) {
        int row = row0 + r + hi*8, col = m + ntile*16;
        float v = accC[ntile][r] + ldg_f(A.bc + col);
        stg_f(A.xf32 + (size_t)row*64 + col, v);
        stg_h(A.xf16 + (size_t)row*64 + col, (_Float16)v);
      }
  }
}

// ---------------------------------------------------------------- q / kt / vt GEMMs
struct QkvArgs {
  const _Float16* xf16;
  const _Float16* WT[5];
  const float*    bias[5];
  _Float16*       out[5];
};

__global__ __launch_bounds__(256) void k_qkv(QkvArgs A) {
  const int lane = threadIdx.x & 31, wv = threadIdx.x >> 5;
  const int m = lane & 15, hi = lane >> 4;
  const int tile = blockIdx.x * 8 + wv;
  if (tile * 16 >= NNODES) return;           // no barriers in this kernel
  const int row0 = tile * 16;
  const _Float16* arow = A.xf16 + (size_t)(row0 + m) * 64;
  const v16h a0 = loadA_g(arow, 0, hi);
  const v16h a1 = loadA_g(arow, 32, hi);
#pragma unroll
  for (int w = 0; w < 5; ++w) {
    v8f acc[4] = {vzero(), vzero(), vzero(), vzero()};
#pragma unroll
    for (int ntile = 0; ntile < 4; ++ntile) {
      const _Float16* br = A.WT[w] + (size_t)(ntile*16 + m) * 64;
      acc[ntile] = wmma16(a0, loadB_h(br, 0, hi),  acc[ntile]);
      acc[ntile] = wmma16(a1, loadB_h(br, 32, hi), acc[ntile]);
    }
#pragma unroll
    for (int ntile = 0; ntile < 4; ++ntile)
#pragma unroll
      for (int r = 0; r < 8; ++r) {
        int row = row0 + r + hi*8, col = m + ntile*16;
        stg_h(A.out[w] + (size_t)row*64 + col,
              (_Float16)(acc[ntile][r] + ldg_f(A.bias[w] + col)));
      }
  }
}

// ---------------------------------------------------------------- edge phase
__global__ __launch_bounds__(256) void k_initneg(float* p, int n) {
  int i = blockIdx.x * blockDim.x + threadIdx.x;
  if (i < n) stg_f(p + i, NEGV);
}

__global__ __launch_bounds__(256) void k_logits(const int* __restrict__ ei,
                                                const int* __restrict__ et,
                                                const _Float16* __restrict__ Q,
                                                const _Float16* __restrict__ KT,
                                                float* __restrict__ logits,
                                                float* __restrict__ MX) {
  int idx = blockIdx.x * blockDim.x + threadIdx.x;   // (edge, head)
  if (idx >= NEDGES * 4) return;
  int e = idx >> 2, h = idx & 3;
  int s = ldg_i(ei + e), d = ldg_i(ei + NEDGES + e), t = ldg_i(et + e);
  const _Float16* qp = Q  + (size_t)d * 64 + h * 16;
  const _Float16* kp = KT + ((size_t)t * NNODES + s) * 64 + h * 16;
  v8h q0 = ldg_v8h(qp), q1 = ldg_v8h(qp + 8);
  v8h k0 = ldg_v8h(kp), k1 = ldg_v8h(kp + 8);
  float acc = 0.f;
#pragma unroll
  for (int i = 0; i < 8; ++i)
    acc += (float)q0[i] * (float)k0[i] + (float)q1[i] * (float)k1[i];
  stg_f(logits + idx, acc);                           // p_rel*scale folded into KT
  atomicMaxF(MX + ((size_t)t * NNODES + d) * 4 + h, acc);
}

__global__ __launch_bounds__(256) void k_expsum(const int* __restrict__ ei,
                                                const int* __restrict__ et,
                                                const float* __restrict__ logits,
                                                const float* __restrict__ MX,
                                                float* __restrict__ DEN,
                                                const _Float16* __restrict__ VT,
                                                float* __restrict__ NUM) {
  int idx = blockIdx.x * blockDim.x + threadIdx.x;
  if (idx >= NEDGES * 4) return;
  int e = idx >> 2, h = idx & 3;
  int s = ldg_i(ei + e), d = ldg_i(ei + NEDGES + e), t = ldg_i(et + e);
  float l  = ldg_f(logits + idx);
  float mx = ldg_f(MX + ((size_t)t * NNODES + d) * 4 + h);
  float ex = expf(l - mx);
  atomicAdd(DEN + ((size_t)t * NNODES + d) * 4 + h, ex);
  const _Float16* vp = VT  + ((size_t)t * NNODES + s) * 64 + h * 16;
  float*          np = NUM + ((size_t)t * NNODES + d) * 64 + h * 16;
  v8h v0 = ldg_v8h(vp), v1 = ldg_v8h(vp + 8);
#pragma unroll
  for (int i = 0; i < 8; ++i) {
    atomicAdd(np + i,     ex * (float)v0[i]);
    atomicAdd(np + 8 + i, ex * (float)v1[i]);
  }
}

// ---------------------------------------------------------------- layer finish
struct FinArgs {
  const float *num, *den;
  const _Float16* WAT;
  const float *ba, *skip;
  float* xf32; _Float16* xf16;
};

__global__ __launch_bounds__(256) void k_finish(FinArgs A) {
  __shared__ _Float16 sh[8][16][72];
  const int lane = threadIdx.x & 31, wv = threadIdx.x >> 5;
  const int m = lane & 15, hi = lane >> 4;
  const int tile = blockIdx.x * 8 + wv;
  const bool act = (tile * 16) < NNODES;
  const int row0 = act ? tile * 16 : 0;
  const float beta = 1.f / (1.f + expf(-ldg_f(A.skip)));

  for (int idx = lane; idx < 16 * 64; idx += 32) {
    int r = idx >> 6, c = idx & 63, h = c >> 4;
    size_t n = (size_t)(row0 + r);
    float v = 0.f;
#pragma unroll
    for (int t = 0; t < 2; ++t) {
      float dn = ldg_f(A.den + ((size_t)t * NNODES + n) * 4 + h);
      float nm = ldg_f(A.num + ((size_t)t * NNODES + n) * 64 + c);
      v += nm / (dn > 0.f ? dn : 1.f);
    }
    float g = 0.5f * v * (1.f + erff(v * 0.70710678118f));   // exact gelu
    sh[wv][r][c] = (_Float16)g;
  }
  __syncthreads();
  v8f acc[4] = {vzero(), vzero(), vzero(), vzero()};
#pragma unroll
  for (int kb = 0; kb < 64; kb += 32) {
    v16h a = loadA_l(&sh[wv][m][0], kb, hi);
#pragma unroll
    for (int ntile = 0; ntile < 4; ++ntile)
      acc[ntile] = wmma16(a, loadB_h(A.WAT + (size_t)(ntile*16 + m) * 64, kb, hi), acc[ntile]);
  }
  if (act) {
#pragma unroll
    for (int ntile = 0; ntile < 4; ++ntile)
#pragma unroll
      for (int r = 0; r < 8; ++r) {
        int row = row0 + r + hi*8, col = m + ntile*16;
        float o  = acc[ntile][r] + ldg_f(A.ba + col);
        float xo = ldg_f(A.xf32 + (size_t)row*64 + col);
        float xn = beta * o + (1.f - beta) * xo;
        xn = xn > 0.f ? xn : 0.f;
        stg_f(A.xf32 + (size_t)row*64 + col, xn);
        stg_h(A.xf16 + (size_t)row*64 + col, (_Float16)xn);
      }
  }
}

// ---------------------------------------------------------------- final head
struct OutArgs {
  const _Float16* xf16; const _Float16* WPT;
  const float *bp, *Wo, *bo; float* out;
};

__global__ __launch_bounds__(256) void k_final(OutArgs A) {
  __shared__ float sh[8][16][68];
  const int lane = threadIdx.x & 31, wv = threadIdx.x >> 5;
  const int m = lane & 15, hi = lane >> 4;
  const int tile = blockIdx.x * 8 + wv;
  const bool act = (tile * 16) < NNODES;
  const int row0 = act ? tile * 16 : 0;

  const _Float16* arow = A.xf16 + (size_t)(row0 + m) * 64;
  const v16h a0 = loadA_g(arow, 0, hi);
  const v16h a1 = loadA_g(arow, 32, hi);
  v8f acc[4] = {vzero(), vzero(), vzero(), vzero()};
#pragma unroll
  for (int ntile = 0; ntile < 4; ++ntile) {
    const _Float16* br = A.WPT + (size_t)(ntile*16 + m) * 64;
    acc[ntile] = wmma16(a0, loadB_h(br, 0, hi),  acc[ntile]);
    acc[ntile] = wmma16(a1, loadB_h(br, 32, hi), acc[ntile]);
  }
#pragma unroll
  for (int ntile = 0; ntile < 4; ++ntile)
#pragma unroll
    for (int r = 0; r < 8; ++r) {
      int rr = r + hi*8, cc = m + ntile*16;
      float v = acc[ntile][r] + ldg_f(A.bp + cc);
      sh[wv][rr][cc] = v > 0.f ? v : 0.f;
    }
  __syncthreads();
  int r = lane & 15, j = lane >> 4;            // lane -> (row, output col)
  float o = ldg_f(A.bo + j);
#pragma unroll
  for (int c = 0; c < 64; ++c) o += sh[wv][r][c] * ldg_f(A.Wo + c*2 + j);
  if (act) stg_f(A.out + (size_t)(row0 + r) * 2 + j, o);
}

// ---------------------------------------------------------------- launcher
extern "C" void kernel_launch(void* const* d_in, const int* in_sizes, int n_in,
                              void* d_out, int out_size, void* d_ws, size_t ws_size,
                              hipStream_t stream) {
  (void)in_sizes; (void)n_in; (void)out_size; (void)ws_size;

  // top-level inputs in setup_inputs() insertion order
  const float* x1 = (const float*)d_in[0];
  const float* x2 = (const float*)d_in[1];
  const float* x3 = (const float*)d_in[2];
  const int*   ei = (const int*)d_in[3];
  const int*   et = (const int*)d_in[4];
  // params leaves in jax tree-flatten (sorted-key) order:
  // Wo, Wp, bo, bp, layers[0..1]{Wa,Wk,Wq,Wv,a_rel,ba,bk,bq,bv,m_rel,p_rel,skip}, mlp{W1,W2,W3,Wc,b1,b2,b3,bc}
  const float* Wo = (const float*)d_in[5];
  const float* Wp = (const float*)d_in[6];
  const float* bo = (const float*)d_in[7];
  const float* bp = (const float*)d_in[8];
  const float *Wa[2], *Wk[2], *Wq[2], *Wv[2], *ar[2], *bal[2], *bk[2], *bq[2], *bv[2], *mr[2], *pr[2], *sk[2];
  for (int L = 0; L < 2; ++L) {
    int b = 9 + L * 12;
    Wa[L]  = (const float*)d_in[b+0];  Wk[L] = (const float*)d_in[b+1];
    Wq[L]  = (const float*)d_in[b+2];  Wv[L] = (const float*)d_in[b+3];
    ar[L]  = (const float*)d_in[b+4];  bal[L] = (const float*)d_in[b+5];
    bk[L]  = (const float*)d_in[b+6];  bq[L] = (const float*)d_in[b+7];
    bv[L]  = (const float*)d_in[b+8];  mr[L] = (const float*)d_in[b+9];
    pr[L]  = (const float*)d_in[b+10]; sk[L] = (const float*)d_in[b+11];
  }
  const float* W1 = (const float*)d_in[33];
  const float* W2 = (const float*)d_in[34];
  const float* W3 = (const float*)d_in[35];
  const float* Wc = (const float*)d_in[36];
  const float* b1 = (const float*)d_in[37];
  const float* b2 = (const float*)d_in[38];
  const float* b3 = (const float*)d_in[39];
  const float* bc = (const float*)d_in[40];

  // ---- workspace carve
  char* wsb = (char*)d_ws;
  size_t off = 0;
  auto alloc = [&](size_t bytes) -> void* {
    void* p = (void*)(wsb + off);
    off += (bytes + 255) & ~(size_t)255;
    return p;
  };
  _Float16* W1T = (_Float16*)alloc((size_t)64*768*2);
  _Float16* W2T = (_Float16*)alloc((size_t)64*768*2);
  _Float16* W3T = (_Float16*)alloc((size_t)64*128*2);
  _Float16* WCT = (_Float16*)alloc((size_t)64*192*2);
  _Float16* WPT = (_Float16*)alloc((size_t)64*64*2);
  _Float16 *WQT[2], *WAT[2], *WKT[2][2], *WVT[2][2];
  float *BKT[2][2], *BVT[2][2];
  for (int L = 0; L < 2; ++L) {
    WQT[L] = (_Float16*)alloc((size_t)64*64*2);
    WAT[L] = (_Float16*)alloc((size_t)64*64*2);
    for (int t = 0; t < 2; ++t) {
      WKT[L][t] = (_Float16*)alloc((size_t)64*64*2);
      WVT[L][t] = (_Float16*)alloc((size_t)64*64*2);
      BKT[L][t] = (float*)alloc((size_t)64*4);
      BVT[L][t] = (float*)alloc((size_t)64*4);
    }
  }
  const size_t N64 = (size_t)NNODES * 64;
  float*    XF32 = (float*)alloc(N64 * 4);
  _Float16* XF16 = (_Float16*)alloc(N64 * 2);
  _Float16* Qb   = (_Float16*)alloc(N64 * 2);
  _Float16* KTb  = (_Float16*)alloc(2 * N64 * 2);
  _Float16* VTb  = (_Float16*)alloc(2 * N64 * 2);
  float*    LOGb = (float*)alloc((size_t)NEDGES * 4 * 4);
  float*    MX   = (float*)alloc((size_t)2 * NNODES * 4 * 4);
  float*    DEN  = (float*)alloc((size_t)2 * NNODES * 4 * 4);
  float*    NUM  = (float*)alloc((size_t)2 * N64 * 4);

  // ---- weight prep
  PrepArgs pa;
  pa.W1 = W1; pa.W2 = W2; pa.W3 = W3; pa.Wc = Wc; pa.Wp = Wp;
  pa.W1T = W1T; pa.W2T = W2T; pa.W3T = W3T; pa.WCT = WCT; pa.WPT = WPT;
  for (int L = 0; L < 2; ++L) {
    pa.Wq[L] = Wq[L]; pa.Wk[L] = Wk[L]; pa.Wv[L] = Wv[L]; pa.Wa[L] = Wa[L];
    pa.ar[L] = ar[L]; pa.mr[L] = mr[L]; pa.pr[L] = pr[L];
    pa.bk[L] = bk[L]; pa.bv[L] = bv[L];
    pa.WQT[L] = WQT[L]; pa.WAT[L] = WAT[L];
    for (int t = 0; t < 2; ++t) {
      pa.WKT[L][t] = WKT[L][t]; pa.WVT[L][t] = WVT[L][t];
      pa.BKT[L][t] = BKT[L][t]; pa.BVT[L][t] = BVT[L][t];
    }
  }
  k_prep<<<dim3(7), dim3(256), 0, stream>>>(pa);

  const int TB = (NNODES/16 + 7) / 8;          // 782 blocks, 8 tiles/block
  const int EB = (NEDGES*4 + 255) / 256;       // edge*head threads

  // ---- input MLP
  MlpArgs ma;
  ma.x1 = x1; ma.x2 = x2; ma.x3 = x3;
  ma.W1T = W1T; ma.W2T = W2T; ma.W3T = W3T; ma.WCT = WCT;
  ma.b1 = b1; ma.b2 = b2; ma.b3 = b3; ma.bc = bc;
  ma.xf32 = XF32; ma.xf16 = XF16;
  k_mlp<<<TB, 256, 0, stream>>>(ma);

  // ---- HGT layers
  for (int L = 0; L < 2; ++L) {
    QkvArgs qa;
    qa.xf16 = XF16;
    qa.WT[0] = WQT[L];    qa.bias[0] = bq[L];     qa.out[0] = Qb;
    qa.WT[1] = WKT[L][0]; qa.bias[1] = BKT[L][0]; qa.out[1] = KTb;
    qa.WT[2] = WKT[L][1]; qa.bias[2] = BKT[L][1]; qa.out[2] = KTb + N64;
    qa.WT[3] = WVT[L][0]; qa.bias[3] = BVT[L][0]; qa.out[3] = VTb;
    qa.WT[4] = WVT[L][1]; qa.bias[4] = BVT[L][1]; qa.out[4] = VTb + N64;
    k_qkv<<<TB, 256, 0, stream>>>(qa);

    hipMemsetAsync(DEN, 0, (size_t)2 * NNODES * 4 * sizeof(float), stream);
    hipMemsetAsync(NUM, 0, (size_t)2 * N64 * sizeof(float), stream);
    k_initneg<<<(2*NNODES*4 + 255)/256, 256, 0, stream>>>(MX, 2*NNODES*4);

    k_logits<<<EB, 256, 0, stream>>>(ei, et, Qb, KTb, LOGb, MX);
    k_expsum<<<EB, 256, 0, stream>>>(ei, et, LOGb, MX, DEN, VTb, NUM);

    FinArgs fa;
    fa.num = NUM; fa.den = DEN; fa.WAT = WAT[L];
    fa.ba = bal[L]; fa.skip = sk[L]; fa.xf32 = XF32; fa.xf16 = XF16;
    k_finish<<<TB, 256, 0, stream>>>(fa);
  }

  // ---- output head
  OutArgs oa;
  oa.xf16 = XF16; oa.WPT = WPT; oa.bp = bp; oa.Wo = Wo; oa.bo = bo;
  oa.out = (float*)d_out;
  k_final<<<TB, 256, 0, stream>>>(oa);
}
